// MultiHeadAttention_64123861729289
// MI455X (gfx1250) — matrix operand order repaired and checked by reading, not compiled, not run on hardware
//
#include <hip/hip_runtime.h>
#include <hip/hip_bf16.h>

typedef __attribute__((ext_vector_type(16))) _Float16 v16h;
typedef __attribute__((ext_vector_type(8)))  float    v8f;
typedef unsigned int __attribute__((ext_vector_type(4))) u32x4;
typedef int  __attribute__((ext_vector_type(4))) i32x4;
typedef int  __attribute__((ext_vector_type(8))) i32x8;

#if __has_builtin(__builtin_amdgcn_tensor_load_to_lds) && \
    __has_builtin(__builtin_amdgcn_s_wait_tensorcnt)
#define USE_TDM 1
#else
#define USE_TDM 0
#endif

// ---------------------------------------------------------------------------
// Helpers
// ---------------------------------------------------------------------------
__device__ __forceinline__ v8f wmma_f16(v16h a, v16h b, v8f c) {
    // D = A(16x32 f16) * B(32x16 f16) + C(16x16 f32)
    return __builtin_amdgcn_wmma_f32_16x16x32_f16(
        /*neg_a=*/false, a, /*neg_b=*/false, b,
        /*c_mod=*/(short)0, c, /*reuse_a=*/false, /*reuse_b=*/false);
}

// A-operand K packing (ISA 16-bit A 16x32 table): vgpr v, lane-half hl
__device__ __forceinline__ int kposA(int v, int hl) {
    return ((v >> 2) << 4) + ((v & 3) << 1) + (hl << 3);
}
// B-operand K packing (same table order as A)
__device__ __forceinline__ int kposB(int v, int hl) {
    return ((v >> 2) << 4) + ((v & 3) << 1) + (hl << 3);
}

__device__ __forceinline__ float rowmax16(float v) {
    v = fmaxf(v, __shfl_xor(v, 1, 32));
    v = fmaxf(v, __shfl_xor(v, 2, 32));
    v = fmaxf(v, __shfl_xor(v, 4, 32));
    v = fmaxf(v, __shfl_xor(v, 8, 32));
    return v;
}
__device__ __forceinline__ float rowsum16(float v) {
    v += __shfl_xor(v, 1, 32);
    v += __shfl_xor(v, 2, 32);
    v += __shfl_xor(v, 4, 32);
    v += __shfl_xor(v, 8, 32);
    return v;
}

#if USE_TDM
// ---------------------------------------------------------------------------
// TDM: async DMA of a 2D f16 tile (tile_w x tile_h, row stride = row_stride
// elements) from global into LDS, padding each 64B row with 16B so the LDS
// row stride becomes 40 halves (matches the fragment-read layout).
// tile_w must be 32 (64B rows) for this pad configuration.
// ---------------------------------------------------------------------------
__device__ __forceinline__ void tdm_load_tile_f16(const _Float16* gsrc,
                                                  unsigned lds_off,
                                                  int tensor_w, int tensor_h,
                                                  int tile_w, int tile_h,
                                                  int row_stride_elems) {
    const unsigned long long ga = (unsigned long long)(uintptr_t)gsrc;
    u32x4 g0;
    g0[0] = 1u;                                  // count=1 (valid), user mode
    g0[1] = lds_off;                             // LDS byte address
    g0[2] = (unsigned)(ga & 0xFFFFFFFFu);        // global_addr[31:0]
    g0[3] = (unsigned)((ga >> 32) & 0x1FFFFFFu)  // global_addr[56:32]
          | (2u << 30);                          // type = 2 ("image")
    i32x8 g1;
    g1[0] = (int)((1u << 16)     // data_size = 1 -> 2 bytes
                | (1u << 20)     // pad_enable
                | (3u << 22)     // pad_interval code 3 -> every 16 DW (64B)
                | (3u << 25));   // pad_amount  code 3 -> 4 DW (16B)
    g1[1] = (int)(((unsigned)tensor_w & 0xFFFFu) << 16);      // tensor_dim0 lo
    g1[2] = (int)((((unsigned)tensor_w >> 16) & 0xFFFFu)      // tensor_dim0 hi
                | (((unsigned)tensor_h & 0xFFFFu) << 16));    // tensor_dim1 lo
    g1[3] = (int)((((unsigned)tensor_h >> 16) & 0xFFFFu)      // tensor_dim1 hi
                | (((unsigned)tile_w & 0xFFFFu) << 16));      // tile_dim0
    g1[4] = (int)((unsigned)tile_h & 0xFFFFu);                // tile_dim1, dim2=0
    g1[5] = row_stride_elems;                                 // dim0_stride lo32
    g1[6] = 0;                                                // stride hi bits
    g1[7] = 0;
    const i32x4 z4 = {0, 0, 0, 0};
#if defined(__clang_major__) && __clang_major__ >= 23
    const i32x8 z8 = {0, 0, 0, 0, 0, 0, 0, 0};
    __builtin_amdgcn_tensor_load_to_lds(g0, g1, z4, z4, z8, 0);
#else
    __builtin_amdgcn_tensor_load_to_lds(g0, g1, z4, z4, 0);
#endif
}
#endif  // USE_TDM

// ---------------------------------------------------------------------------
// f32 -> f16 conversion (grid-stride)
// ---------------------------------------------------------------------------
__global__ void cvt_f32_to_f16(const float* __restrict__ in,
                               _Float16* __restrict__ out, int n) {
    int i = blockIdx.x * blockDim.x + threadIdx.x;
    int stride = gridDim.x * blockDim.x;
    for (; i < n; i += stride) out[i] = (_Float16)in[i];
}

// ---------------------------------------------------------------------------
// Tiled WMMA GEMM:  C[M,N] = A[M,K](f16) * B[K,N](f16) + bias[N](f32)
// Block = 256 threads (8 waves). Workgroup tile 128x128, wave tile 64x32.
// Software-pipelined, double-buffered LDS:
//   - A tile i+1 streamed by the Tensor Data Mover (TENSORcnt) while tile i
//     is consumed by WMMAs (manual register staging as fallback).
//   - B tile i+1 pre-loaded to registers during compute, stored (transposed)
//     to LDS at the top of the next iteration.
// ---------------------------------------------------------------------------
template <bool OUT_HALF>
__global__ __launch_bounds__(256)
void wmma_gemm(const _Float16* __restrict__ A, const _Float16* __restrict__ Bw,
               const float* __restrict__ bias, void* __restrict__ Cout,
               int M, int N, int K) {
    __shared__ _Float16 As[2][128 * 40];  // A tiles, row-major, stride 40
    __shared__ _Float16 Bt[2][128 * 40];  // B tiles transposed: Bt[n][k]

    const int tid    = threadIdx.x;
    const int wv     = tid >> 5;
    const int lane   = tid & 31;
    const int lanelo = lane & 15;
    const int hl     = lane >> 4;

    const int N0 = blockIdx.x * 128;
    const int M0 = blockIdx.y * 128;
    const int wm0 = (wv >> 2) * 64;   // wave M offset in tile
    const int wn0 = (wv & 3) * 32;    // wave N offset in tile

    const v8f vzero = {0.f, 0.f, 0.f, 0.f, 0.f, 0.f, 0.f, 0.f};
    v8f acc[4][2];
    for (int mt = 0; mt < 4; ++mt)
        for (int nt = 0; nt < 2; ++nt) acc[mt][nt] = vzero;

    const int nk = K >> 5;  // K-steps of 32

    // B register-staging geometry: thread covers 16 n at one k-row
    const int bk  = tid >> 3;         // k row within tile: 0..31
    const int bnb = (tid & 7) * 16;   // n base within tile
    union U8 { uint4 u; _Float16 h[8]; };
    U8 b0, b1;  // pre-fetched B chunk for the upcoming tile

#if !USE_TDM
    const int am  = tid >> 1;         // A staging: row 0..127
    const int akb = (tid & 1) * 16;   // A staging: k base 0 or 16
    uint4 a0, a1;
#endif

    // ---------------- prologue: start tile 0 ----------------
#if USE_TDM
    if (wv == 0) {
        tdm_load_tile_f16(A + (size_t)M0 * K, (unsigned)(uintptr_t)As[0],
                          K, M - M0, 32, 128, K);
    }
#else
    {
        const uint4* src = (const uint4*)(A + (size_t)(M0 + am) * K + akb);
        a0 = src[0];
        a1 = src[1];
    }
#endif
    {
        const _Float16* src = Bw + (size_t)bk * N + N0 + bnb;
        b0.u = *(const uint4*)src;
        b1.u = *(const uint4*)(src + 8);
    }

    for (int i = 0; i < nk; ++i) {
        const int p = i & 1;

        // (a) commit pre-fetched B (and A, fallback) registers to LDS buffer p
        {
            _Float16* bt = Bt[p];
            for (int j = 0; j < 8; ++j) bt[(bnb + j) * 40 + bk]     = b0.h[j];
            for (int j = 0; j < 8; ++j) bt[(bnb + 8 + j) * 40 + bk] = b1.h[j];
        }
#if !USE_TDM
        {
            _Float16* as = As[p];
            *(uint4*)(as + am * 40 + akb)     = a0;
            *(uint4*)(as + am * 40 + akb + 8) = a1;
        }
#endif

        // (b) TDM for tile i has fully landed in As[p]
#if USE_TDM
        if (wv == 0) __builtin_amdgcn_s_wait_tensorcnt(0);
#endif
        // (c) publish buffer p; also guarantees all waves finished reading
        //     buffer p^1 in iteration i-1, so it is safe to refill it
        __syncthreads();

        // (d,e) kick off tile i+1 into buffer p^1 while we compute on tile i
        if (i + 1 < nk) {
            const int kt1 = (i + 1) * 32;
#if USE_TDM
            if (wv == 0) {
                tdm_load_tile_f16(A + (size_t)M0 * K + kt1,
                                  (unsigned)(uintptr_t)As[p ^ 1],
                                  K - kt1, M - M0, 32, 128, K);
            }
#else
            {
                const uint4* src =
                    (const uint4*)(A + (size_t)(M0 + am) * K + kt1 + akb);
                a0 = src[0];
                a1 = src[1];
            }
#endif
            const _Float16* src = Bw + (size_t)(kt1 + bk) * N + N0 + bnb;
            b0.u = *(const uint4*)src;
            b1.u = *(const uint4*)(src + 8);
        }

        // (f) build fragments from LDS buffer p and issue WMMAs
        v16h af[4], bf[2];
        const _Float16* asb = As[p];
        const _Float16* btb = Bt[p];
        for (int mt = 0; mt < 4; ++mt) {
            const _Float16* ap = asb + (wm0 + mt * 16 + lanelo) * 40;
            for (int v = 0; v < 8; ++v) {
                int kp = kposA(v, hl);
                af[mt][2 * v]     = ap[kp];
                af[mt][2 * v + 1] = ap[kp + 1];
            }
        }
        for (int nt = 0; nt < 2; ++nt) {
            const _Float16* bp = btb + (wn0 + nt * 16 + lanelo) * 40;
            for (int v = 0; v < 8; ++v) {
                int kp = kposB(v, hl);
                bf[nt][2 * v]     = bp[kp];
                bf[nt][2 * v + 1] = bp[kp + 1];
            }
        }
        for (int mt = 0; mt < 4; ++mt)
            for (int nt = 0; nt < 2; ++nt)
                acc[mt][nt] = wmma_f16(af[mt], bf[nt], acc[mt][nt]);
    }

    // ---- epilogue: bias add + store (D layout: row = r + 8*hl, col = lane%16)
    for (int mt = 0; mt < 4; ++mt) {
        for (int nt = 0; nt < 2; ++nt) {
            const int col = N0 + wn0 + nt * 16 + lanelo;
            const float bv = bias[col];
            for (int r = 0; r < 8; ++r) {
                const int row = M0 + wm0 + mt * 16 + r + 8 * hl;
                const float val = acc[mt][nt][r] + bv;
                if constexpr (OUT_HALF)
                    ((_Float16*)Cout)[(size_t)row * N + col] = (_Float16)val;
                else
                    ((float*)Cout)[(size_t)row * N + col] = val;
            }
        }
    }
}

// ---------------------------------------------------------------------------
// Flash-style causal attention.
// qkv layout (from GEMM1): f16 [B*T, 3072]; per (t,h): q @ h*192, k @ +64, v @ +128
// Grid: x = B*H (64), y = T/64 (32). Block = 128 threads (4 waves x 16 q rows).
// Output: f16 [B*T, 1024] at column h*64.
// ---------------------------------------------------------------------------
#define ATT_T   2048
#define ATT_C3  3072
#define ATT_C   1024

__global__ __launch_bounds__(128)
void flash_attn_wmma(const _Float16* __restrict__ qkv,
                     _Float16* __restrict__ outh) {
    __shared__ _Float16 Vt[64 * 40];       // V^T tile: Vt[d][k], 32 keys
    __shared__ _Float16 Pw[4 * 16 * 40];   // per-wave P buffer (16 x 32, pad 40)

    const int tid    = threadIdx.x;
    const int wave   = tid >> 5;
    const int lane   = tid & 31;
    const int lanelo = lane & 15;
    const int hl     = lane >> 4;

    const int bh = blockIdx.x;
    const int b  = bh >> 4;
    const int h  = bh & 15;
    const int q0 = blockIdx.y * 64;
    const int qrowBase = q0 + wave * 16;

    const _Float16* qkvb = qkv + (size_t)b * ATT_T * ATT_C3;

    // ---- Q fragments (A operand), d = 0..31 and 32..63, resident all loop ----
    v16h aq0, aq1;
    {
        const _Float16* qp =
            qkvb + (size_t)(qrowBase + lanelo) * ATT_C3 + h * 192;
        for (int v = 0; v < 8; ++v) {
            int kp = kposA(v, hl);
            aq0[2 * v]     = qp[kp];
            aq0[2 * v + 1] = qp[kp + 1];
            aq1[2 * v]     = qp[32 + kp];
            aq1[2 * v + 1] = qp[32 + kp + 1];
        }
    }

    const v8f vzero = {0.f, 0.f, 0.f, 0.f, 0.f, 0.f, 0.f, 0.f};
    v8f o[4];
    for (int dt = 0; dt < 4; ++dt) o[dt] = vzero;
    float rmax[8], rsum[8];
    for (int r = 0; r < 8; ++r) { rmax[r] = -1e30f; rsum[r] = 0.f; }

    const float scale = 0.125f;  // 1/sqrt(64)
    const int kbEnd = (q0 + 64) / 32;  // causal: key blocks of 32

    _Float16* pw = Pw + wave * 16 * 40;

    for (int kb = 0; kb < kbEnd; ++kb) {
        const int kbase = kb * 32;
        __syncthreads();  // previous iteration's LDS reads complete

        // ---- stage V block transposed: Vt[d][k] (32 keys x 64 d) ----
        {
            const int k     = tid >> 2;       // 0..31
            const int dbase = (tid & 3) * 16; // 0,16,32,48
            const _Float16* src =
                qkvb + (size_t)(kbase + k) * ATT_C3 + h * 192 + 128 + dbase;
            for (int j = 0; j < 16; ++j) Vt[(dbase + j) * 40 + k] = src[j];
        }

        // ---- S = Q*K^T for 32 keys (two 16-key sub-tiles) ----
        v8f s[2];
        for (int sb = 0; sb < 2; ++sb) {
            v16h bk0, bk1;
            const _Float16* kp_ =
                qkvb + (size_t)(kbase + sb * 16 + lanelo) * ATT_C3 + h * 192 + 64;
            for (int v = 0; v < 8; ++v) {
                int kp = kposB(v, hl);
                bk0[2 * v]     = kp_[kp];
                bk0[2 * v + 1] = kp_[kp + 1];
                bk1[2 * v]     = kp_[32 + kp];
                bk1[2 * v + 1] = kp_[32 + kp + 1];
            }
            v8f sacc = vzero;
            sacc = wmma_f16(aq0, bk0, sacc);
            sacc = wmma_f16(aq1, bk1, sacc);
            // causal mask + scale (D layout: row = r + 8*hl, key = col)
            const int key = kbase + sb * 16 + lanelo;
            for (int r = 0; r < 8; ++r) {
                const int qr = qrowBase + r + 8 * hl;
                s[sb][r] = (key <= qr) ? sacc[r] * scale : -1e30f;
            }
        }

        // ---- online softmax across these 32 keys ----
        float scl[8];
        for (int r = 0; r < 8; ++r) {
            float m = rowmax16(fmaxf(s[0][r], s[1][r]));
            float mnew = fmaxf(rmax[r], m);
            float sc   = __expf(rmax[r] - mnew);
            float p0   = __expf(s[0][r] - mnew);
            float p1   = __expf(s[1][r] - mnew);
            s[0][r] = p0;
            s[1][r] = p1;
            rsum[r] = rsum[r] * sc + rowsum16(p0 + p1);
            rmax[r] = mnew;
            scl[r]  = sc;
        }
        for (int dt = 0; dt < 4; ++dt)
            for (int r = 0; r < 8; ++r) o[dt][r] *= scl[r];

        // ---- P: D-layout -> LDS -> A-layout fragment ----
        for (int sb = 0; sb < 2; ++sb)
            for (int r = 0; r < 8; ++r)
                pw[(r + 8 * hl) * 40 + sb * 16 + lanelo] = (_Float16)s[sb][r];
        __syncthreads();  // Vt staged + P visible

        v16h ap;
        {
            const _Float16* prow = pw + lanelo * 40;
            for (int v = 0; v < 8; ++v) {
                int kp = kposA(v, hl);
                ap[2 * v]     = prow[kp];
                ap[2 * v + 1] = prow[kp + 1];
            }
        }
        // ---- O += P * V  (K = 32 keys, 4 d-tiles of 16) ----
        for (int dt = 0; dt < 4; ++dt) {
            v16h bv;
            const _Float16* vrow = Vt + (dt * 16 + lanelo) * 40;
            for (int v = 0; v < 8; ++v) {
                int kp = kposB(v, hl);
                bv[2 * v]     = vrow[kp];
                bv[2 * v + 1] = vrow[kp + 1];
            }
            o[dt] = wmma_f16(ap, bv, o[dt]);
        }
    }

    // ---- normalize and store f16 [B*T, 1024] at column h*64 ----
    for (int dt = 0; dt < 4; ++dt) {
        for (int r = 0; r < 8; ++r) {
            const int qr  = qrowBase + r + 8 * hl;
            const int col = h * 64 + dt * 16 + lanelo;
            const float val = o[dt][r] / rsum[r];
            outh[(size_t)(b * ATT_T + qr) * ATT_C + col] = (_Float16)val;
        }
    }
}

// ---------------------------------------------------------------------------
// Launch: cvt -> QKV GEMM -> attention -> proj GEMM
// ---------------------------------------------------------------------------
extern "C" void kernel_launch(void* const* d_in, const int* in_sizes, int n_in,
                              void* d_out, int out_size, void* d_ws,
                              size_t ws_size, hipStream_t stream) {
    const float* x     = (const float*)d_in[0];  // [4,2048,1024]
    const float* Wqkv  = (const float*)d_in[1];  // [1024,3072]
    const float* bqkv  = (const float*)d_in[2];  // [3072]
    const float* Wproj = (const float*)d_in[3];  // [1024,1024]
    const float* bproj = (const float*)d_in[4];  // [1024]
    float* out = (float*)d_out;                  // [4,2048,1024] f32

    char* ws = (char*)d_ws;
    _Float16* xh     = (_Float16*)(ws);                 // 8192*1024 f16 (16 MB)
    _Float16* wqkvh  = (_Float16*)(ws + 16777216);      // 1024*3072 f16 ( 6 MB)
    _Float16* wprojh = (_Float16*)(ws + 23068672);      // 1024*1024 f16 ( 2 MB)
    _Float16* qkvh   = (_Float16*)(ws + 25165824);      // 8192*3072 f16 (48 MB)
    _Float16* attnh  = (_Float16*)(ws + 75497472);      // 8192*1024 f16 (16 MB)

    cvt_f32_to_f16<<<2048, 256, 0, stream>>>(x,     xh,     8192 * 1024);
    cvt_f32_to_f16<<<2048, 256, 0, stream>>>(Wqkv,  wqkvh,  1024 * 3072);
    cvt_f32_to_f16<<<1024, 256, 0, stream>>>(Wproj, wprojh, 1024 * 1024);

    // qkv = x @ W_qkv + b_qkv   (M=8192, N=3072, K=1024) -> f16
    wmma_gemm<true><<<dim3(24, 64), 256, 0, stream>>>(
        xh, wqkvh, bqkv, (void*)qkvh, 8192, 3072, 1024);

    // causal flash attention per (b,h,q-block)
    flash_attn_wmma<<<dim3(64, 32), 128, 0, stream>>>(qkvh, attnh);

    // out = attn @ W_proj + b_proj  (M=8192, N=1024, K=1024) -> f32
    wmma_gemm<false><<<dim3(8, 64), 256, 0, stream>>>(
        attnh, wprojh, bproj, (void*)out, 8192, 1024, 1024);
}
